// LSTM_42090679501192
// MI455X (gfx1250) — matrix operand order repaired and checked by reading, not compiled, not run on hardware
//
#include <hip/hip_runtime.h>
#include <hip/hip_bf16.h>

// ---------------- problem constants ----------------
#define BB     64
#define LL     512
#define UU     7
#define HH     128
#define GG     512            // 4*H
#define NSEQ   (BB*UU)        // 448 sequences (raw reshape)
#define MROWS  (NSEQ*LL)      // 229376 flat rows
#define PRED   96

typedef __attribute__((ext_vector_type(16))) _Float16 v16h;
typedef __attribute__((ext_vector_type(8)))  _Float16 v8h;
typedef __attribute__((ext_vector_type(8)))  float    v8f;

union V16 { v16h v; v8h h2[2]; };

// A fragment: 16x32 (MxK) f16, row-major source, row stride ld halves.
// lane<16: halves0-7 = K0..7, halves8-15 = K16..23 of row M=lane
// lane>=16: halves0-7 = K8..15, halves8-15 = K24..31 of row M=lane-16
__device__ __forceinline__ v16h load_a_frag(const _Float16* base, int ld, int lane) {
    int r = lane & 15;
    const _Float16* p = base + r * ld + ((lane & 16) ? 8 : 0);
    V16 u;
    u.h2[0] = *(const v8h*)(p);
    u.h2[1] = *(const v8h*)(p + 16);
    return u.v;
}

// B fragment for D = A*B with B = W^T (gates = h @ W^T):
// B[k][n] = W[col0+n][k].  lane n<16: halves = W[col0+n][k0..k0+7, k0+16..k0+23]
//                          lane>=16: halves = W[col0+n][k0+8..k0+15, k0+24..k0+31]
__device__ __forceinline__ v16h load_b_frag(const _Float16* base, int ld, int lane) {
    int n = lane & 15;
    const _Float16* p = base + n * ld + ((lane & 16) ? 8 : 0);
    V16 u;
    u.h2[0] = *(const v8h*)(p);
    u.h2[1] = *(const v8h*)(p + 16);
    return u.v;
}

__device__ __forceinline__ float sigm_(float x) { return 1.f / (1.f + __expf(-x)); }
__device__ __forceinline__ float tanh_(float x) { return 2.f / (1.f + __expf(-2.f * x)) - 1.f; }

// CDNA5 async DMA: stage a 512x128 f16 weight panel (128KB) global -> LDS.
// Uses GLOBAL_LOAD_ASYNC_TO_LDS_B128 (ASYNCcnt-tracked, no VGPR bounce).
__device__ __forceinline__ void stage_weights_async(const _Float16* g, _Float16* s) {
    unsigned sbase = (unsigned)(unsigned long long)(void*)s; // low 32b = LDS offset
    unsigned long long gbase = (unsigned long long)g;
#pragma unroll 1
    for (int i = threadIdx.x; i < GG * HH / 8; i += 256) {   // 8192 x 16B chunks
        unsigned lds = sbase + (unsigned)i * 16u;
        unsigned long long ga = gbase + (unsigned long long)i * 16ull;
        asm volatile("global_load_async_to_lds_b128 %0, %1, off"
                     :: "v"(lds), "v"(ga) : "memory");
    }
    asm volatile("s_wait_asynccnt 0x0" ::: "memory");
    __syncthreads();
}

// ------------- kernel 0: weight convert + bias fuse -------------
__global__ void k_convert(const float* __restrict__ Wih, const float* __restrict__ Whh,
                          const float* __restrict__ bih, const float* __restrict__ bhh,
                          _Float16* __restrict__ Wih16, _Float16* __restrict__ Whh16,
                          float* __restrict__ bsum) {
    int i = blockIdx.x * 256 + threadIdx.x;           // grid covers 2*512*128
    if (i < 2 * GG * HH) {
        Wih16[i] = (_Float16)Wih[i];
        Whh16[i] = (_Float16)Whh[i];
    }
    if (i < 2 * GG) bsum[i] = bih[i] + bhh[i];
}

// ------------- kernel 1: per-(b,u) instance-norm stats -------------
__global__ void k_stats(const float* __restrict__ x, float* __restrict__ mean,
                        float* __restrict__ stdv) {
    int bu = blockIdx.x;                               // 448 blocks
    int b = bu / UU, u = bu % UU;
    const float* xb = x + (long)b * LL * UU + u;
    float s = 0.f, s2 = 0.f;
    for (int l = threadIdx.x; l < LL; l += 256) {
        float v = xb[(long)l * UU];
        s += v; s2 += v * v;
    }
    __shared__ float r1[256], r2[256];
    r1[threadIdx.x] = s; r2[threadIdx.x] = s2;
    __syncthreads();
    for (int off = 128; off > 0; off >>= 1) {
        if (threadIdx.x < off) {
            r1[threadIdx.x] += r1[threadIdx.x + off];
            r2[threadIdx.x] += r2[threadIdx.x + off];
        }
        __syncthreads();
    }
    if (threadIdx.x == 0) {
        float m = r1[0] / (float)LL;
        float var = r2[0] / (float)LL - m * m;
        mean[bu] = m;
        stdv[bu] = sqrtf(var + 1e-5f);
    }
}

// ------------- kernel 2: normalize (raw flat reshape) + embed to f16 -------------
__global__ void k_embed(const float* __restrict__ x, const float* __restrict__ mean,
                        const float* __restrict__ stdv, const float* __restrict__ embW,
                        const float* __restrict__ embb, _Float16* __restrict__ h0) {
    __shared__ float sx[256];
    __shared__ float sw[HH], sb[HH];
    long i0 = (long)blockIdx.x * 256;                  // 896 blocks
    {
        long i = i0 + threadIdx.x;                     // flat over [B,L,U]
        int b = (int)(i / (LL * UU));
        int u = (int)(i % UU);
        int bu = b * UU + u;
        sx[threadIdx.x] = (x[i] - mean[bu]) / stdv[bu];
    }
    if (threadIdx.x < HH) {
        sw[threadIdx.x] = embW[threadIdx.x];
        sb[threadIdx.x] = embb[threadIdx.x];
    }
    __syncthreads();
    _Float16* out = h0 + i0 * HH;
    for (int k = 0; k < HH; k++) {
        int idx = threadIdx.x + k * 256;
        int il = idx >> 7, hh = idx & 127;
        out[idx] = (_Float16)(sx[il] * sw[hh] + sb[hh]);
    }
}

// ------------- kernel 3: xp = hin @ Wih^T + bias  (WMMA GEMM) -------------
__global__ void k_xp_gemm(const _Float16* __restrict__ hin, const _Float16* __restrict__ W,
                          const float* __restrict__ bias, _Float16* __restrict__ xp) {
    extern __shared__ char smem[];
    _Float16* sW = (_Float16*)smem;                    // 512*128 f16 = 128KB
    stage_weights_async(W, sW);

    int wave = threadIdx.x >> 5, lane = threadIdx.x & 31;
    long row0 = (long)blockIdx.x * 128 + wave * 16;    // 1792 blocks, 8 waves

    const _Float16* A = hin + row0 * HH;
    v16h afrag[4];
#pragma unroll
    for (int k = 0; k < 4; k++) afrag[k] = load_a_frag(A + k * 32, HH, lane);

    for (int nt = 0; nt < 32; ++nt) {                  // 32 gate tiles of 16
        int gc0 = nt * 16;
        v8f acc = {};
#pragma unroll
        for (int k = 0; k < 4; k++) {
            v16h b = load_b_frag(sW + gc0 * HH + k * 32, HH, lane);
            acc = __builtin_amdgcn_wmma_f32_16x16x32_f16(false, afrag[k], false, b,
                                                         (short)0, acc, false, false);
        }
        int n = lane & 15;
        int mb = (lane & 16) ? 8 : 0;
        float bc = bias[gc0 + n];
#pragma unroll
        for (int r = 0; r < 8; r++)
            xp[(row0 + mb + r) * GG + gc0 + n] = (_Float16)(acc[r] + bc);
    }
}

// ------------- kernel 4: persistent recurrent scan (per 16 sequences) -------------
// Per step critical path: barrier -> 4 LDS A-frag loads -> 16 WMMA -> barrier ->
// fused cell update with xp prefetched one step ahead (latency hidden).
__global__ void k_recur(const _Float16* __restrict__ xp, const _Float16* __restrict__ W,
                        _Float16* __restrict__ hout) {
    extern __shared__ char smem[];
    _Float16* sW     = (_Float16*)smem;                       // 128KB
    float*    sGates = (float*)(smem + GG * HH * 2);          // 16*512 f32 = 32KB
    _Float16* sH     = (_Float16*)(smem + GG * HH * 2 + 16 * GG * 4); // 16*128 f16

    for (int i = threadIdx.x; i < 16 * HH; i += 256) sH[i] = (_Float16)0.f;
    stage_weights_async(W, sW);   // includes __syncthreads()

    int wave = threadIdx.x >> 5, lane = threadIdx.x & 31;
    int n0 = blockIdx.x * 16;                          // 28 blocks

    // Hoist B fragments (this wave's 64 gate columns) into registers: 16 frags
    v16h bfrag[4][4];
#pragma unroll
    for (int nt = 0; nt < 4; nt++)
#pragma unroll
        for (int k = 0; k < 4; k++)
            bfrag[nt][k] = load_b_frag(sW + (wave * 64 + nt * 16) * HH + k * 32, HH, lane);

    // Elementwise ownership: m = tid>>4 (row), j0 = (tid&15)*8 (8 consecutive cols)
    int m  = threadIdx.x >> 4;
    int j0 = (threadIdx.x & 15) << 3;
    float creg[8];
#pragma unroll
    for (int r = 0; r < 8; r++) creg[r] = 0.f;

    long xbase = ((long)(n0 + m) * LL) * GG + j0;      // xp row origin for t=0
    long hbase = ((long)(n0 + m) * LL) * HH + j0;

    v8h xq[4];
    {
        const _Float16* p = xp + xbase;
        xq[0] = *(const v8h*)(p);
        xq[1] = *(const v8h*)(p + 128);
        xq[2] = *(const v8h*)(p + 256);
        xq[3] = *(const v8h*)(p + 384);
    }

    for (int t = 0; t < LL; t++) {
        __syncthreads();                               // h (prev step) visible
        v16h afrag[4];
#pragma unroll
        for (int k = 0; k < 4; k++) afrag[k] = load_a_frag(sH + k * 32, HH, lane);

        // prefetch xp for next step (overlaps WMMA phase)
        int tn = (t + 1 < LL) ? (t + 1) : t;
        v8h xqn[4];
        {
            const _Float16* p = xp + xbase + (long)tn * GG;
            xqn[0] = *(const v8h*)(p);
            xqn[1] = *(const v8h*)(p + 128);
            xqn[2] = *(const v8h*)(p + 256);
            xqn[3] = *(const v8h*)(p + 384);
        }

#pragma unroll
        for (int nt = 0; nt < 4; nt++) {
            v8f acc = {};
#pragma unroll
            for (int k = 0; k < 4; k++)
                acc = __builtin_amdgcn_wmma_f32_16x16x32_f16(false, afrag[k], false,
                                                             bfrag[nt][k], (short)0, acc,
                                                             false, false);
            int n = lane & 15;
            int mb = (lane & 16) ? 8 : 0;
            int gc = wave * 64 + nt * 16 + n;
#pragma unroll
            for (int r = 0; r < 8; r++) sGates[(mb + r) * GG + gc] = acc[r];
        }
        __syncthreads();                               // gates visible

        const float* gr = sGates + m * GG + j0;
        v8h hv;
#pragma unroll
        for (int i = 0; i < 8; i++) {
            float gi = gr[i]            + (float)xq[0][i];
            float gf = gr[HH + i]       + (float)xq[1][i];
            float gg = gr[2 * HH + i]   + (float)xq[2][i];
            float go = gr[3 * HH + i]   + (float)xq[3][i];
            float iv = sigm_(gi), fv = sigm_(gf), gv = tanh_(gg), ov = sigm_(go);
            float c = fv * creg[i] + iv * gv;
            creg[i] = c;
            hv[i] = (_Float16)(ov * tanh_(c));
        }
        *(v8h*)(sH + m * HH + j0) = hv;                      // ds b128
        *(v8h*)(hout + hbase + (long)t * HH) = hv;           // global b128
#pragma unroll
        for (int q = 0; q < 4; q++) xq[q] = xqn[q];
    }
}

// ------------- kernel 5: FC head + denorm scatter -------------
__global__ void k_fc(const _Float16* __restrict__ hfinal, const float* __restrict__ fcW,
                     const float* __restrict__ fcb, const float* __restrict__ mean,
                     const float* __restrict__ stdv, float* __restrict__ out) {
    int idx = blockIdx.x * 256 + threadIdx.x;          // 448*96
    if (idx >= NSEQ * PRED) return;
    int n = idx / PRED, p = idx % PRED;
    const _Float16* h = hfinal + ((long)n * LL + (LL - 1)) * HH;
    const float* w = fcW + p * HH;
    float acc = fcb[p];
    for (int k = 0; k < HH; k++) acc += (float)h[k] * w[k];
    // out flat = [B*U, PRED] reshaped flat to [B, PRED, U]; 672 % 7 == 0
    int f = idx;
    int b = f / (PRED * UU);
    int u = f % UU;
    out[f] = acc * stdv[b * UU + u] + mean[b * UU + u];
}

// ---------------- host launcher ----------------
extern "C" void kernel_launch(void* const* d_in, const int* in_sizes, int n_in,
                              void* d_out, int out_size, void* d_ws, size_t ws_size,
                              hipStream_t stream) {
    const float* x     = (const float*)d_in[0];
    const float* embW  = (const float*)d_in[1];
    const float* embb  = (const float*)d_in[2];
    const float* Wih   = (const float*)d_in[3];
    const float* Whh   = (const float*)d_in[4];
    const float* bih   = (const float*)d_in[5];
    const float* bhh   = (const float*)d_in[6];
    const float* fcW   = (const float*)d_in[7];
    const float* fcb   = (const float*)d_in[8];
    float* out = (float*)d_out;

    char* ws = (char*)d_ws;
    size_t off = 0;
    auto alloc = [&](size_t bytes) { size_t r = off; off += (bytes + 255) & ~(size_t)255; return r; };
    _Float16* Wih16 = (_Float16*)(ws + alloc(2 * GG * HH * sizeof(_Float16)));
    _Float16* Whh16 = (_Float16*)(ws + alloc(2 * GG * HH * sizeof(_Float16)));
    float*    bsum  = (float*)   (ws + alloc(2 * GG * sizeof(float)));
    float*    mean  = (float*)   (ws + alloc(NSEQ * sizeof(float)));
    float*    stdv  = (float*)   (ws + alloc(NSEQ * sizeof(float)));
    _Float16* hA    = (_Float16*)(ws + alloc((size_t)MROWS * HH * sizeof(_Float16)));
    _Float16* hB    = (_Float16*)(ws + alloc((size_t)MROWS * HH * sizeof(_Float16)));
    _Float16* xpb   = (_Float16*)(ws + alloc((size_t)MROWS * GG * sizeof(_Float16)));

    const int LDS_GEMM  = GG * HH * 2;                       // 128KB
    const int LDS_RECUR = GG * HH * 2 + 16 * GG * 4 + 16 * HH * 2; // 164KB
    (void)hipFuncSetAttribute((const void*)k_xp_gemm,
                              hipFuncAttributeMaxDynamicSharedMemorySize, LDS_GEMM);
    (void)hipFuncSetAttribute((const void*)k_recur,
                              hipFuncAttributeMaxDynamicSharedMemorySize, LDS_RECUR);

    k_convert<<<(2 * GG * HH + 255) / 256, 256, 0, stream>>>(Wih, Whh, bih, bhh,
                                                             Wih16, Whh16, bsum);
    k_stats<<<NSEQ, 256, 0, stream>>>(x, mean, stdv);
    k_embed<<<MROWS / 256, 256, 0, stream>>>(x, mean, stdv, embW, embb, hA);

    for (int layer = 0; layer < 2; layer++) {
        const _Float16* hin = layer ? hB : hA;
        _Float16* ho = layer ? hA : hB;
        k_xp_gemm<<<MROWS / 128, 256, LDS_GEMM, stream>>>(hin, Wih16 + layer * GG * HH,
                                                          bsum + layer * GG, xpb);
        k_recur<<<NSEQ / 16, 256, LDS_RECUR, stream>>>(xpb, Whh16 + layer * GG * HH, ho);
    }
    k_fc<<<(NSEQ * PRED + 255) / 256, 256, 0, stream>>>(hA, fcW, fcb, mean, stdv, out);
}
